// BahdanauAttention_71906342470035
// MI455X (gfx1250) — matrix operand order repaired and checked by reading, not compiled, not run on hardware
//
#include <hip/hip_runtime.h>
#include <hip/hip_bf16.h>
#include <cmath>

typedef __attribute__((ext_vector_type(16))) __bf16 v16bf;
typedef __attribute__((ext_vector_type(8)))  float  v8f;

#define BB 64
#define SS 2048
#define DD 512
#define UU 512
#define SB 520   // LDS A-tile row pitch in bf16 elements (1040 B: rows shift 4 banks -> conflict-free b128)

__device__ __forceinline__ unsigned short f2bf(float x) {
    unsigned int u = __float_as_uint(x);
    u += 0x7FFFu + ((u >> 16) & 1u);      // round-to-nearest-even
    return (unsigned short)(u >> 16);
}

union AFrag { uint4 q[2]; v16bf v; };

// ---------------------------------------------------------------------------
// Kernel 1: base[b,u] = dec_hidden[b,:] @ W2[:,u] + b1[u] + b2[u] + bc[u]
// ---------------------------------------------------------------------------
__global__ __launch_bounds__(256) void base_kernel(
    const float* __restrict__ dh, const float* __restrict__ W2,
    const float* __restrict__ b1, const float* __restrict__ b2,
    const float* __restrict__ bc, float* __restrict__ basev)
{
    const int b = blockIdx.y;
    const int u = blockIdx.x * 256 + threadIdx.x;
    float acc = b1[u] + b2[u] + bc[u];
    const float* dhr = dh + b * DD;
    #pragma unroll 4
    for (int k = 0; k < DD; ++k)
        acc = fmaf(dhr[k], W2[k * UU + u], acc);
    basev[b * UU + u] = acc;
}

// ---------------------------------------------------------------------------
// Kernel 2: repack W1 (fp32 [D,U]) into bf16 WMMA B-fragment order.
// Flat index i = (((t*16 + kk)*32 + lane)*16 + e)
//   k = kk*32 + (lane/16)*16 + e ; n = t*16 + (lane%16)
// ---------------------------------------------------------------------------
__global__ __launch_bounds__(256) void pack_kernel(
    const float* __restrict__ W1, unsigned short* __restrict__ wpack)
{
    const int i  = blockIdx.x * 256 + threadIdx.x;   // < 32*16*32*16 = 262144
    const int e  = i & 15;
    const int l  = (i >> 4) & 31;
    const int kk = (i >> 9) & 15;
    const int t  = i >> 13;
    const int k  = kk * 32 + (l >> 4) * 16 + e;
    const int n  = t * 16 + (l & 15);
    wpack[i] = f2bf(W1[k * UU + n]);
}

// ---------------------------------------------------------------------------
// Kernel 3 (main): per 32-row M-block (two 16-row M-tiles share every B
// fragment), WMMA GEMM enc_output@W1 fused with tanh(C + base + pcov*Wc).Vw
// -> e[b,s].  8 waves; each wave: 4 N-tiles x 2 M-tiles = 8 accumulators.
// ---------------------------------------------------------------------------
__global__ __launch_bounds__(256) void attn_scores_kernel(
    const float* __restrict__ enc,            // [B,S,D] fp32
    const unsigned short* __restrict__ wpack, // packed bf16 W1 (512 KB, L2-resident)
    const float* __restrict__ basev,          // [B,U]
    const float* __restrict__ pcov,           // [B,S]
    const float* __restrict__ Wc,             // [U]
    const float* __restrict__ Vw,             // [U]
    const float* __restrict__ bv,             // [1]
    float* __restrict__ e_out)                // [B,S]
{
    __shared__ unsigned short ldsA[32 * SB];  // 33280 B
    __shared__ float pcov_s[32];
    __shared__ float e_sm[32];

    const int tid = threadIdx.x;
    const int wg  = blockIdx.x;        // B * S/32 = 4096
    const int b   = wg >> 6;
    const int s0  = (wg & 63) << 5;

    // ---- stage A block (32 x 512 fp32 -> bf16) into LDS, coalesced float4 ----
    const float* arow = enc + ((size_t)b * SS + s0) * DD;
    #pragma unroll
    for (int j = 0; j < 16; ++j) {
        const int idx = tid + j * 256;        // float4 units, 0..4095
        const int row = idx >> 7;             // /128
        const int c4  = idx & 127;
        const float4 v = ((const float4*)(arow + row * DD))[c4];
        unsigned short* dst = &ldsA[row * SB + c4 * 4];
        dst[0] = f2bf(v.x); dst[1] = f2bf(v.y);
        dst[2] = f2bf(v.z); dst[3] = f2bf(v.w);
    }
    if (tid < 32) {
        pcov_s[tid] = pcov[(size_t)b * SS + s0 + tid];
        e_sm[tid]   = 0.0f;
    }
    __syncthreads();

    const int wave = tid >> 5;
    const int lane = tid & 31;
    const int half = lane >> 4;
    const int lm   = lane & 15;

    v8f acc[2][4];
    #pragma unroll
    for (int mt = 0; mt < 2; ++mt)
        #pragma unroll
        for (int j = 0; j < 4; ++j)
            #pragma unroll
            for (int r = 0; r < 8; ++r) acc[mt][j][r] = 0.0f;

    // 32-bit invariant offsets (wpack < 512 KB; ldsA < 64 KB)
    const int boff  = wave * 4 * 8192 + lane * 16;  // bf16-element offset, j=0 kk=0
    const int arow0 = lm * SB + half * 16;          // M-tile 0; tile 1 at +16*SB

    // ---- K loop: 16 steps; per step: 4 ds_b128 (A) + 8 glb_b128 (B) + 8 WMMA --
    for (int kk = 0; kk < 16; ++kk) {
        AFrag af[2];
        const int col = arow0 + kk * 32;
        af[0].q[0] = *(const uint4*)&ldsA[col];
        af[0].q[1] = *(const uint4*)&ldsA[col + 8];
        af[1].q[0] = *(const uint4*)&ldsA[col + 16 * SB];
        af[1].q[1] = *(const uint4*)&ldsA[col + 16 * SB + 8];
        #pragma unroll
        for (int j = 0; j < 4; ++j) {
            const v16bf bfrag = *(const v16bf*)(wpack + (boff + j * 8192 + kk * 512));
            acc[0][j] = __builtin_amdgcn_wmma_f32_16x16x32_bf16(
                false, af[0].v, false, bfrag, (short)0, acc[0][j], false, false);
            acc[1][j] = __builtin_amdgcn_wmma_f32_16x16x32_bf16(
                false, af[1].v, false, bfrag, (short)0, acc[1][j], false, false);
        }
    }

    // ---- fused epilogue: tanh + dot(Vw), lane/tile/wave reduce ----
    float eacc[2][8];
    #pragma unroll
    for (int mt = 0; mt < 2; ++mt)
        #pragma unroll
        for (int r = 0; r < 8; ++r) eacc[mt][r] = 0.0f;

    #pragma unroll
    for (int j = 0; j < 4; ++j) {
        const int n    = (wave * 4 + j) * 16 + lm;
        const float bn  = basev[b * UU + n];
        const float wcn = Wc[n];
        const float vwn = Vw[n];
        #pragma unroll
        for (int mt = 0; mt < 2; ++mt) {
            #pragma unroll
            for (int r = 0; r < 8; ++r) {
                const int m = mt * 16 + half * 8 + r;
                const float feat = tanhf(acc[mt][j][r] + bn + pcov_s[m] * wcn);
                eacc[mt][r] = fmaf(feat, vwn, eacc[mt][r]);
            }
        }
    }
    #pragma unroll
    for (int mt = 0; mt < 2; ++mt)
        #pragma unroll
        for (int r = 0; r < 8; ++r) {
            float v = eacc[mt][r];
            v += __shfl_xor(v, 1, 32);
            v += __shfl_xor(v, 2, 32);
            v += __shfl_xor(v, 4, 32);
            v += __shfl_xor(v, 8, 32);   // reduced within each 16-lane half
            eacc[mt][r] = v;
        }
    if (lm == 0) {
        #pragma unroll
        for (int mt = 0; mt < 2; ++mt)
            #pragma unroll
            for (int r = 0; r < 8; ++r)
                atomicAdd(&e_sm[mt * 16 + half * 8 + r], eacc[mt][r]); // ds_add_f32
    }
    __syncthreads();
    if (tid < 32)
        e_out[(size_t)b * SS + s0 + tid] = e_sm[tid] + bv[0];
}

// ---------------------------------------------------------------------------
// Kernel 4: masked softmax + renormalize + coverage.  One block per batch row.
// ---------------------------------------------------------------------------
__global__ __launch_bounds__(256) void softmax_kernel(
    const float* __restrict__ e, const int* __restrict__ mask,
    const float* __restrict__ pcov,
    float* __restrict__ attn_out, float* __restrict__ cov_out)
{
    __shared__ float red[256];
    const int b = blockIdx.x, tid = threadIdx.x;
    const float* eb = e + (size_t)b * SS;

    float vals[8];
    float mx = -INFINITY;
    #pragma unroll
    for (int j = 0; j < 8; ++j) {
        vals[j] = eb[tid + j * 256];
        mx = fmaxf(mx, vals[j]);
    }
    red[tid] = mx; __syncthreads();
    for (int off = 128; off > 0; off >>= 1) {
        if (tid < off) red[tid] = fmaxf(red[tid], red[tid + off]);
        __syncthreads();
    }
    mx = red[0]; __syncthreads();

    // p*mask then renormalize (global softmax denominator cancels)
    float msum = 0.0f;
    float mk[8];
    #pragma unroll
    for (int j = 0; j < 8; ++j) {
        vals[j] = expf(vals[j] - mx);
        mk[j] = (float)mask[(size_t)b * SS + tid + j * 256];
        msum += vals[j] * mk[j];
    }
    red[tid] = msum; __syncthreads();
    for (int off = 128; off > 0; off >>= 1) {
        if (tid < off) red[tid] += red[tid + off];
        __syncthreads();
    }
    const float inv = 1.0f / red[0];

    #pragma unroll
    for (int j = 0; j < 8; ++j) {
        const size_t idx = (size_t)b * SS + tid + j * 256;
        const float a = vals[j] * mk[j] * inv;
        attn_out[idx] = a;
        cov_out[idx]  = a + pcov[idx];
    }
}

// ---------------------------------------------------------------------------
// Kernel 5: context[b,d] = sum_s attn[b,s] * enc[b,s,d]   (bandwidth-bound)
// ---------------------------------------------------------------------------
__global__ __launch_bounds__(256) void context_kernel(
    const float* __restrict__ enc, const float* __restrict__ attn,
    float* __restrict__ ctx)
{
    __shared__ float a_s[256];
    const int b = blockIdx.y;
    const int d = blockIdx.x * 256 + threadIdx.x;
    float acc = 0.0f;
    for (int s0 = 0; s0 < SS; s0 += 256) {
        __syncthreads();
        a_s[threadIdx.x] = attn[(size_t)b * SS + s0 + threadIdx.x];
        __syncthreads();
        const float* ep = enc + ((size_t)b * SS + s0) * DD + d;
        #pragma unroll 4
        for (int s = 0; s < 256; ++s)
            acc = fmaf(a_s[s], ep[s * DD], acc);
    }
    ctx[b * DD + d] = acc;
}

// ---------------------------------------------------------------------------
extern "C" void kernel_launch(void* const* d_in, const int* in_sizes, int n_in,
                              void* d_out, int out_size, void* d_ws, size_t ws_size,
                              hipStream_t stream) {
    const float* dec_hidden = (const float*)d_in[0];   // [B,H]
    const float* enc_output = (const float*)d_in[1];   // [B,S,D]
    const float* pre_cov    = (const float*)d_in[2];   // [B,S,1]
    const int*   pad_mask   = (const int*)  d_in[3];   // [B,S]
    const float* W1 = (const float*)d_in[4];
    const float* b1 = (const float*)d_in[5];
    const float* W2 = (const float*)d_in[6];
    const float* b2 = (const float*)d_in[7];
    const float* Wc = (const float*)d_in[8];
    const float* bc = (const float*)d_in[9];
    const float* Vw = (const float*)d_in[10];
    const float* bv = (const float*)d_in[11];

    float* out  = (float*)d_out;
    float* ctx  = out;                       // [B,D]   = 32768
    float* attn = out + BB * DD;             // [B,S]   = 131072
    float* cov  = out + BB * DD + BB * SS;   // [B,S,1] = 131072

    // workspace: base[B*U] f32 | e[B*S] f32 | wpack[32*16*32*16] bf16  (~1.13 MB)
    float* basev = (float*)d_ws;
    float* e_ws  = basev + BB * UU;
    unsigned short* wpack = (unsigned short*)(e_ws + BB * SS);

    base_kernel<<<dim3(UU / 256, BB), 256, 0, stream>>>(dec_hidden, W2, b1, b2, bc, basev);
    pack_kernel<<<(32 * 16 * 32 * 16) / 256, 256, 0, stream>>>(W1, wpack);
    attn_scores_kernel<<<BB * (SS / 32), 256, 0, stream>>>(
        enc_output, wpack, basev, pre_cov, Wc, Vw, bv, e_ws);
    softmax_kernel<<<BB, 256, 0, stream>>>(e_ws, pad_mask, pre_cov, attn, cov);
    context_kernel<<<dim3(DD / 256, BB), 256, 0, stream>>>(enc_output, attn, ctx);
}
